// ImpellerLayer_14499809591535
// MI455X (gfx1250) — matrix operand order repaired and checked by reading, not compiled, not run on hardware
//
#include <hip/hip_runtime.h>

typedef __attribute__((ext_vector_type(16))) _Float16 v16h;
typedef __attribute__((ext_vector_type(8)))  float    v8f;

#define HIDDEN   128
#define NUM_PATH 8
#define PATH_LEN 4
#define K_TOT    256   // 2 * HIDDEN
#define TILE_M   16

__global__ __launch_bounds__(256)
void impeller_fused_kernel(const float* __restrict__ feats,
                           const long long* __restrict__ paths,
                           const long long* __restrict__ path_types,
                           const float* __restrict__ path_weights,
                           const float* __restrict__ W_fc,
                           float* __restrict__ out,
                           int N, int numTiles)
{
    __shared__ int      idx_lds[TILE_M * NUM_PATH * PATH_LEN];   // 2 KB
    __shared__ _Float16 fout_lds[TILE_M][K_TOT];                 // 8 KB

    const int t    = threadIdx.x;
    const int lane = t & 31;
    const int wave = t >> 5;          // 0..7 -> output column tile

    // ---- path types / weights (wave-uniform -> SGPRs)
    int   tp[NUM_PATH];
    float wv[NUM_PATH][PATH_LEN];
    float cnt0 = 0.f, cnt1 = 0.f;
#pragma unroll
    for (int p = 0; p < NUM_PATH; ++p) {
        tp[p] = (int)path_types[p];
        cnt0 += (tp[p] == 0) ? 1.f : 0.f;
        cnt1 += (tp[p] == 0) ? 0.f : 1.f;
#pragma unroll
        for (int l = 0; l < PATH_LEN; ++l)
            wv[p][l] = path_weights[tp[p] * PATH_LEN + l];
    }
    const float inv0 = 1.f / cnt0;
    const float inv1 = 1.f / cnt1;
    // fold edge-type select + mean scaling into per-path scalars (SGPRs):
    // a0 += s0[p]*c; a1 += s1[p]*c  ->  one v_fmac each instead of cndmask+add
    float s0[NUM_PATH], s1[NUM_PATH];
#pragma unroll
    for (int p = 0; p < NUM_PATH; ++p) {
        s0[p] = (tp[p] == 0) ? inv0 : 0.f;
        s1[p] = (tp[p] == 0) ? 0.f : inv1;
    }

    // ---- preload B operand (this wave's 16 output cols of W_fc, f16), reused all tiles
    // B k-order used here (contiguous, same as A): lane: n = lane%16, koff = lane<16 ? 0 : 16;
    //                           VGPR v holds K = kbase + koff + 2v, +1
    v16h Bv[8];
    {
        const int n    = lane & 15;
        const int koff = (lane < 16) ? 0 : 16;
        const float* wrow = W_fc + (size_t)(wave * 16 + n) * K_TOT;
#pragma unroll
        for (int c = 0; c < 8; ++c) {
            union { v16h v; _Float16 h[16]; } u;
#pragma unroll
            for (int v = 0; v < 8; ++v) {
                const int k = 32 * c + koff + 2 * v;
                u.h[2 * v]     = (_Float16)wrow[k];
                u.h[2 * v + 1] = (_Float16)wrow[k + 1];
            }
            Bv[c] = u.v;
        }
    }

    // ---- phase-1 thread mapping: lane -> feature quad, wave -> node group
    const int q = lane;               // feature quad id: features [4q .. 4q+3]
    const int g = wave;               // node group: handles nodes {g, g+8}
    const float4* fquad = (const float4*)feats;   // 32 quads per node row

    for (int tile = blockIdx.x; tile < numTiles; tile += gridDim.x) {
        const int node0 = tile * TILE_M;

        __syncthreads();   // protect LDS reuse across iterations

        // ---- stage all 16*8*4 = 512 path indices into LDS
        for (int j = t; j < TILE_M * NUM_PATH * PATH_LEN; j += 256) {
            const int n = j >> 5;          // node in tile
            const int p = (j >> 2) & 7;    // path
            const int l = j & 3;           // hop
            int nn = node0 + n;
            if (nn >= N) nn = N - 1;
            idx_lds[j] = (int)paths[(size_t)(p * N + nn) * PATH_LEN + l];
        }
        __syncthreads();

        // ---- phase 1: gather (b128, batched for MLP) + per-type weighted mean
#pragma unroll
        for (int half = 0; half < 2; ++half) {
            const int n  = g + 8 * half;                 // node in tile
            const int* ib = idx_lds + n * (NUM_PATH * PATH_LEN);
            float a0x = 0.f, a0y = 0.f, a0z = 0.f, a0w = 0.f;
            float a1x = 0.f, a1y = 0.f, a1z = 0.f, a1w = 0.f;
#pragma unroll
            for (int pp = 0; pp < NUM_PATH; pp += 4) {
                // batch 16 b128 loads before any use -> deep load pipelining
                float4 f[4][4];
#pragma unroll
                for (int p = 0; p < 4; ++p)
#pragma unroll
                    for (int l = 0; l < 4; ++l)
                        f[p][l] = fquad[ib[(pp + p) * PATH_LEN + l] * 32 + q];
#pragma unroll
                for (int p = 0; p < 4; ++p) {
                    float cx = 0.f, cy = 0.f, cz = 0.f, cw = 0.f;
#pragma unroll
                    for (int l = 0; l < 4; ++l) {
                        const float w = wv[pp + p][l];
                        cx += w * f[p][l].x;
                        cy += w * f[p][l].y;
                        cz += w * f[p][l].z;
                        cw += w * f[p][l].w;
                    }
                    const float w0 = s0[pp + p];
                    const float w1 = s1[pp + p];
                    a0x += w0 * cx;  a1x += w1 * cx;
                    a0y += w0 * cy;  a1y += w1 * cy;
                    a0z += w0 * cz;  a1z += w1 * cz;
                    a0w += w0 * cw;  a1w += w1 * cw;
                }
            }
            // pack 4 f16 and store as one b64 per type (mean scaling already folded)
            union { _Float16 h[4]; unsigned long long u; } pk0, pk1;
            pk0.h[0] = (_Float16)a0x; pk0.h[1] = (_Float16)a0y;
            pk0.h[2] = (_Float16)a0z; pk0.h[3] = (_Float16)a0w;
            pk1.h[0] = (_Float16)a1x; pk1.h[1] = (_Float16)a1y;
            pk1.h[2] = (_Float16)a1z; pk1.h[3] = (_Float16)a1w;
            *(unsigned long long*)&fout_lds[n][4 * q]          = pk0.u;
            *(unsigned long long*)&fout_lds[n][HIDDEN + 4 * q] = pk1.u;
        }
        __syncthreads();

        // ---- phase 2: 8 K-chunks of WMMA, A streamed from LDS
        // A k-order used here (contiguous, same as B): lane: m = lane%16, klo = lane<16 ? 0 : 16;
        //   VGPR v: K = kbase + klo + 2*v, +1 (packed pair)
        v8f acc = {};
        {
            const int m   = lane & 15;
            const int klo = (lane < 16) ? 0 : 16;
            const unsigned* row = (const unsigned*)&fout_lds[m][0];
#pragma unroll
            for (int c = 0; c < 8; ++c) {
                union { v16h v; unsigned u[8]; } a;
#pragma unroll
                for (int v = 0; v < 8; ++v) {
                    const int k = 32 * c + klo + 2 * v;
                    a.u[v] = row[k >> 1];
                }
                acc = __builtin_amdgcn_wmma_f32_16x16x32_f16(
                        false, a.v, false, Bv[c], (short)0, acc, false, false);
            }
        }

        // ---- ReLU + store. C/D layout: lane: n = lane%16, M = v + (lane<16 ? 0 : 8)
        {
            const int n     = lane & 15;
            const int mbase = (lane < 16) ? 0 : 8;
            float* obase = out + (size_t)node0 * HIDDEN + wave * 16 + n
                               + (size_t)mbase * HIDDEN;
            if (node0 + TILE_M <= N) {
                // full tile: straight-line stores with immediate offsets
#pragma unroll
                for (int v = 0; v < 8; ++v)
                    obase[(size_t)v * HIDDEN] = fmaxf(acc[v], 0.f);
            } else {
#pragma unroll
                for (int v = 0; v < 8; ++v) {
                    if (node0 + mbase + v < N)
                        obase[(size_t)v * HIDDEN] = fmaxf(acc[v], 0.f);
                }
            }
        }
    }
}

extern "C" void kernel_launch(void* const* d_in, const int* in_sizes, int n_in,
                              void* d_out, int out_size, void* d_ws, size_t ws_size,
                              hipStream_t stream) {
    const float*     feats        = (const float*)d_in[0];
    const long long* paths        = (const long long*)d_in[1];
    const long long* path_types   = (const long long*)d_in[2];
    const float*     path_weights = (const float*)d_in[3];
    const float*     W_fc         = (const float*)d_in[4];
    float*           out          = (float*)d_out;

    const int N        = in_sizes[0] / HIDDEN;           // 100000
    const int numTiles = (N + TILE_M - 1) / TILE_M;      // 6250
    int grid = numTiles < 1250 ? numTiles : 1250;        // each block ~5 tiles, amortizes B preload

    impeller_fused_kernel<<<grid, 256, 0, stream>>>(
        feats, paths, path_types, path_weights, W_fc, out, N, numTiles);
}
